// ContextQueryAttention_57114475102291
// MI455X (gfx1250) — matrix operand order repaired and checked by reading, not compiled, not run on hardware
//
#include <hip/hip_runtime.h>
#include <hip/hip_bf16.h>

// Problem constants (match reference)
#define BB 32
#define CC 2048
#define QQ 128
#define DD 128
#define MASKNEG (-1e30f)

typedef __attribute__((ext_vector_type(16))) __bf16 v16bf;
typedef __attribute__((ext_vector_type(8)))  float  v8f;

#if __has_builtin(__builtin_amdgcn_global_load_async_to_lds_b128) && \
    __has_builtin(__builtin_amdgcn_s_wait_asynccnt)
#define HAVE_ASYNC_LDS 1
typedef int v4i __attribute__((vector_size(16)));
typedef __attribute__((address_space(1))) v4i gv4i;   // global-AS int4
typedef __attribute__((address_space(3))) v4i lv4i;   // LDS-AS int4
#endif

union BFU { unsigned short u; __bf16 h; };

static __device__ __forceinline__ unsigned short f2bf(float f) {
    union { float f; unsigned u; } v; v.f = f;
    unsigned r = v.u + 0x7FFFu + ((v.u >> 16) & 1u);   // round-to-nearest-even
    return (unsigned short)(r >> 16);
}
static __device__ __forceinline__ __bf16 us2bf(unsigned short u) {
    BFU x; x.u = u; return x.h;
}

static __device__ __forceinline__ v8f wmma_bf16(v16bf a, v16bf b, v8f c) {
    return __builtin_amdgcn_wmma_f32_16x16x32_bf16(
        /*neg_a=*/false, a, /*neg_b=*/false, b,
        /*c_mod=*/(short)0, c, /*reuse_a=*/false, /*reuse_b=*/false);
}

// A-fragment (16x32 bf16, contiguous k order): S is [row][col] bf16(ushort), row stride lda.
// lane 0-15 hold M=lane; K = e + ((lane>>4)<<4)
static __device__ __forceinline__ v16bf ldA(const unsigned short* S, int lda,
                                            int m0, int k0, int lane) {
    int M  = lane & 15;
    int kb = (lane >> 4) << 4;
    v16bf a;
#pragma unroll
    for (int e = 0; e < 16; ++e) {
        int K = e + kb;
        a[e] = us2bf(S[(m0 + M) * lda + (k0 + K)]);
    }
    return a;
}
// A-fragment gathered from a transposed store: A[M,K] = S[K][M]
static __device__ __forceinline__ v16bf ldA_T(const unsigned short* S, int lda,
                                              int m0, int k0, int lane) {
    int M  = lane & 15;
    int kb = (lane >> 4) << 4;
    v16bf a;
#pragma unroll
    for (int e = 0; e < 16; ++e) {
        int K = e + kb;
        a[e] = us2bf(S[(k0 + K) * lda + (m0 + M)]);
    }
    return a;
}
// B-fragment (32x16 bf16): N = lane&15; K = e + ((lane>>4)<<4).
// Source stored [k][n] with row stride lda.
static __device__ __forceinline__ v16bf ldB_kn(const unsigned short* S, int lda,
                                               int k0, int n0, int lane) {
    int N  = lane & 15;
    int kb = (lane >> 4) << 4;
    v16bf b;
#pragma unroll
    for (int e = 0; e < 16; ++e) {
        int K = e + kb;
        b[e] = us2bf(S[(k0 + K) * lda + (n0 + N)]);
    }
    return b;
}
// Source stored [n][k] (e.g. row-major x_query[q][d] used as B[d][q])
static __device__ __forceinline__ v16bf ldB_nk(const unsigned short* S, int lda,
                                               int k0, int n0, int lane) {
    int N  = lane & 15;
    int kb = (lane >> 4) << 4;
    v16bf b;
#pragma unroll
    for (int e = 0; e < 16; ++e) {
        int K = e + kb;
        b[e] = us2bf(S[(n0 + N) * lda + (k0 + K)]);
    }
    return b;
}

// ---------------------------------------------------------------------------
// K1: sim[b,c,q] = xc·(xq*W2)^T + s0[q] + s1[c] + bias, masked.  grid(16, B)x256
// ---------------------------------------------------------------------------
__global__ __launch_bounds__(256) void k_sim(
    const float* __restrict__ xc, const float* __restrict__ xq,
    const unsigned char* __restrict__ cmask, const unsigned char* __restrict__ qmask,
    const float* __restrict__ W0, const float* __restrict__ W1,
    const float* __restrict__ W2, const float* __restrict__ bias,
    float* __restrict__ sim)
{
    extern __shared__ char smem[];
    unsigned short* Xc = (unsigned short*)smem;        // [128][136] bf16 x_context tile
    unsigned short* Xq = Xc + 128 * 136;               // [128][136] bf16 (x_query * W2), [q][d]
    float* s0row = (float*)(Xq + 128 * 136);           // [128] per-q
    float* s1col = s0row + 128;                        // [128] per-c
    unsigned char* qm = (unsigned char*)(s1col + 128); // [128]
    unsigned char* cm = qm + 128;                      // [128]

    const int b = blockIdx.y, c0 = blockIdx.x * 128, tid = threadIdx.x;

    for (int i = tid; i < 128 * 128; i += 256) {
        int r = i >> 7, c = i & 127;
        Xq[r * 136 + c] = f2bf(xq[((size_t)b * QQ + r) * DD + c] * W2[c]);
        Xc[r * 136 + c] = f2bf(xc[((size_t)b * CC + c0 + r) * DD + c]);
    }
    if (tid < 128) { qm[tid] = qmask[b * QQ + tid]; cm[tid] = cmask[(size_t)b * CC + c0 + tid]; }
    __syncthreads();

    if (tid < 128) {               // s0[q] = xq[q,:]·W0  (f32 precision)
        float acc = 0.f; const float* p = &xq[((size_t)b * QQ + tid) * DD];
        for (int d = 0; d < DD; ++d) acc += p[d] * W0[d];
        s0row[tid] = acc;
    } else {                       // s1[c] = xc[c,:]·W1
        int r = tid - 128; float acc = 0.f;
        const float* p = &xc[((size_t)b * CC + c0 + r) * DD];
        for (int d = 0; d < DD; ++d) acc += p[d] * W1[d];
        s1col[r] = acc;
    }
    __syncthreads();

    const int w = tid >> 5, lane = tid & 31, m0 = w * 16;
    v8f acc[8];
#pragma unroll
    for (int nt = 0; nt < 8; ++nt)
#pragma unroll
        for (int r = 0; r < 8; ++r) acc[nt][r] = 0.f;

    for (int kk = 0; kk < DD; kk += 32) {
        v16bf a = ldA(Xc, 136, m0, kk, lane);
#pragma unroll
        for (int nt = 0; nt < 8; ++nt) {
            v16bf bf = ldB_nk(Xq, 136, kk, nt * 16, lane);  // B[d][q] from Xq[q][d]
            acc[nt] = wmma_bf16(a, bf, acc[nt]);
        }
    }
    const float bb = bias[0];
#pragma unroll
    for (int nt = 0; nt < 8; ++nt) {
        int n = nt * 16 + (lane & 15);
#pragma unroll
        for (int r = 0; r < 8; ++r) {
            int m = m0 + r + ((lane >> 4) << 3);
            float v = acc[nt][r] + s0row[n] + s1col[m] + bb;
            int ok = (int)cm[m] & (int)qm[n];               // branchless -> v_cndmask
            sim[((size_t)b * CC + c0 + m) * QQ + n] = ok ? v : MASKNEG;
        }
    }
}

// ---------------------------------------------------------------------------
// K2: per-(b,q) online max/sum-exp over c (2048).  grid(B) x 1024
// ---------------------------------------------------------------------------
__global__ __launch_bounds__(1024) void k_colstats(
    const float* __restrict__ sim, float* __restrict__ colmax, float* __restrict__ colsum)
{
    __shared__ float Ms[8][128];
    __shared__ float Ss[8][128];
    const int b = blockIdx.x, t = threadIdx.x, q = t & 127, seg = t >> 7;
    float m = -3.0e38f, s = 0.f;
    for (int c = seg * 256; c < seg * 256 + 256; ++c) {
        float x = sim[((size_t)b * CC + c) * QQ + q];
        if (x > m) { s = s * __expf(m - x) + 1.f; m = x; }
        else       { s += __expf(x - m); }
    }
    Ms[seg][q] = m; Ss[seg][q] = s;
    __syncthreads();
    if (seg == 0) {
#pragma unroll
        for (int k = 1; k < 8; ++k) {
            float m2 = Ms[k][q], s2 = Ss[k][q];
            float M = fmaxf(m, m2);
            s = s * __expf(m - M) + s2 * __expf(m2 - M);
            m = M;
        }
        colmax[b * QQ + q] = m;
        colsum[b * QQ + q] = s;
    }
}

// ---------------------------------------------------------------------------
// K3: row softmax (over q) -> act (bf16, ws) ; c2q = act @ xq.  grid(16,B)x256
// ---------------------------------------------------------------------------
__global__ __launch_bounds__(256) void k_rowsoft_c2q(
    const float* __restrict__ sim, const float* __restrict__ xq,
    unsigned short* __restrict__ actout, float* __restrict__ c2q_out)
{
    extern __shared__ char smem[];
    float*          Sf = (float*)smem;                 // [128][132] sim tile
    unsigned short* Aa = (unsigned short*)(Sf + 128 * 132); // [128][136] c2q_act bf16
    unsigned short* Xq = Aa + 128 * 136;               // [128][136] xq bf16 [q][d]

    const int b = blockIdx.y, c0 = blockIdx.x * 128, tid = threadIdx.x;

#ifdef HAVE_ASYNC_LDS
    // sim tile is a pure f32 copy: use async global->LDS b128 (no VGPR round-trip)
    for (int j = tid; j < 128 * 32; j += 256) {
        int r = j >> 5, cq = (j & 31) * 4;
        const float* g = &sim[((size_t)b * CC + c0 + r) * QQ + cq];
        float*       l = &Sf[r * 132 + cq];
        __builtin_amdgcn_global_load_async_to_lds_b128((gv4i*)g, (lv4i*)l, 0, 0);
    }
    for (int i = tid; i < 128 * 128; i += 256) {
        int r = i >> 7, c = i & 127;
        Xq[r * 136 + c] = f2bf(xq[((size_t)b * QQ + r) * DD + c]);
    }
    __builtin_amdgcn_s_wait_asynccnt(0);
#else
    for (int i = tid; i < 128 * 128; i += 256) {
        int r = i >> 7, c = i & 127;
        Sf[r * 132 + c] = sim[((size_t)b * CC + c0 + r) * QQ + c];
        Xq[r * 136 + c] = f2bf(xq[((size_t)b * QQ + r) * DD + c]);
    }
#endif
    __syncthreads();

    if (tid < 128) {                      // full row fits in tile: exact softmax
        int r = tid;
        float m = -3.0e38f;
        for (int q = 0; q < QQ; ++q) m = fmaxf(m, Sf[r * 132 + q]);
        float s = 0.f;
        for (int q = 0; q < QQ; ++q) s += __expf(Sf[r * 132 + q] - m);
        float rs = 1.f / s;
        for (int q = 0; q < QQ; ++q)
            Aa[r * 136 + q] = f2bf(__expf(Sf[r * 132 + q] - m) * rs);
    }
    __syncthreads();

    for (int i = tid; i < 128 * 128; i += 256) {       // persist act for K5
        int r = i >> 7, c = i & 127;
        actout[((size_t)b * CC + c0 + r) * QQ + c] = Aa[r * 136 + c];
    }

    const int w = tid >> 5, lane = tid & 31, m0 = w * 16;
    v8f acc[8];
#pragma unroll
    for (int nt = 0; nt < 8; ++nt)
#pragma unroll
        for (int r = 0; r < 8; ++r) acc[nt][r] = 0.f;

    for (int kk = 0; kk < QQ; kk += 32) {              // M=c, K=q, N=d
        v16bf a = ldA(Aa, 136, m0, kk, lane);
#pragma unroll
        for (int nt = 0; nt < 8; ++nt) {
            v16bf bf = ldB_kn(Xq, 136, kk, nt * 16, lane);  // B[q][d]
            acc[nt] = wmma_bf16(a, bf, acc[nt]);
        }
    }
#pragma unroll
    for (int nt = 0; nt < 8; ++nt) {
        int n = nt * 16 + (lane & 15);
#pragma unroll
        for (int r = 0; r < 8; ++r) {
            int m = m0 + r + ((lane >> 4) << 3);
            c2q_out[((size_t)b * CC + c0 + m) * DD + n] = acc[nt][r];
        }
    }
}

// ---------------------------------------------------------------------------
// K4: tmp[b,q,d] = sum_c q2c_act[c,q] * xc[c,d]; K=2048 streamed, VGPR accum
// (deterministic, no atomics).  grid(B) x 256
// ---------------------------------------------------------------------------
__global__ __launch_bounds__(256) void k_tmp(
    const float* __restrict__ sim, const float* __restrict__ xc,
    const float* __restrict__ colmax, const float* __restrict__ colsum,
    float* __restrict__ tmp)
{
    extern __shared__ char smem[];
    unsigned short* Ab  = (unsigned short*)smem;       // [32][136] q2c_act chunk [c][q]
    unsigned short* Xcc = Ab + 32 * 136;               // [32][136] xc chunk [c][d]
    float* cmL = (float*)(Xcc + 32 * 136);             // [128] colmax
    float* rsL = cmL + 128;                            // [128] 1/colsum

    const int b = blockIdx.x, tid = threadIdx.x;
    if (tid < 128) { cmL[tid] = colmax[b * QQ + tid]; rsL[tid] = 1.f / colsum[b * QQ + tid]; }
    __syncthreads();

    const int w = tid >> 5, lane = tid & 31, m0 = w * 16;  // M = q rows
    v8f acc[8];
#pragma unroll
    for (int nt = 0; nt < 8; ++nt)
#pragma unroll
        for (int r = 0; r < 8; ++r) acc[nt][r] = 0.f;

    for (int cc = 0; cc < CC; cc += 32) {
        // prefetch next chunk (gfx1250 global_prefetch_b8)
        if (cc + 32 < CC) {
            int pr = tid >> 3, pc = (tid & 7) * 16;
            __builtin_prefetch(&sim[((size_t)b * CC + cc + 32 + pr) * QQ + pc], 0, 3);
            __builtin_prefetch(&xc [((size_t)b * CC + cc + 32 + pr) * DD + pc], 0, 3);
        }
        // load 32x128 sim chunk -> q2c_act bf16, and 32x128 xc chunk -> bf16
        for (int i = tid; i < 32 * 128; i += 256) {
            int r = i >> 7, c = i & 127;
            float x = sim[((size_t)b * CC + cc + r) * QQ + c];
            Ab[r * 136 + c]  = f2bf(__expf(x - cmL[c]) * rsL[c]);
            Xcc[r * 136 + c] = f2bf(xc[((size_t)b * CC + cc + r) * DD + c]);
        }
        __syncthreads();
        v16bf a = ldA_T(Ab, 136, m0, 0, lane);         // A[q, c_local] = Ab[c_local][q]
#pragma unroll
        for (int nt = 0; nt < 8; ++nt) {
            v16bf bf = ldB_kn(Xcc, 136, 0, nt * 16, lane);  // B[c_local][d]
            acc[nt] = wmma_bf16(a, bf, acc[nt]);
        }
        __syncthreads();
    }
#pragma unroll
    for (int nt = 0; nt < 8; ++nt) {
        int n = nt * 16 + (lane & 15);
#pragma unroll
        for (int r = 0; r < 8; ++r) {
            int m = m0 + r + ((lane >> 4) << 3);
            tmp[((size_t)b * QQ + m) * DD + n] = acc[nt][r];
        }
    }
}

// ---------------------------------------------------------------------------
// K5: q2c[b,c,d] = act @ tmp.  grid(16,B) x 256
// ---------------------------------------------------------------------------
__global__ __launch_bounds__(256) void k_q2c(
    const unsigned short* __restrict__ act, const float* __restrict__ tmp,
    float* __restrict__ q2c_out)
{
    extern __shared__ char smem[];
    unsigned short* Aa = (unsigned short*)smem;        // [128][136] act tile [c][q]
    unsigned short* Tm = Aa + 128 * 136;               // [128][136] tmp bf16 [q][d]

    const int b = blockIdx.y, c0 = blockIdx.x * 128, tid = threadIdx.x;

#ifdef HAVE_ASYNC_LDS
    // act tile is already bf16 in ws: pure copy -> async global->LDS b128
    for (int j = tid; j < 128 * 16; j += 256) {
        int r = j >> 4, cq = (j & 15) * 8;             // 8 ushorts = 16B
        const unsigned short* g = &act[((size_t)b * CC + c0 + r) * QQ + cq];
        unsigned short*       l = &Aa[r * 136 + cq];
        __builtin_amdgcn_global_load_async_to_lds_b128((gv4i*)g, (lv4i*)l, 0, 0);
    }
    for (int i = tid; i < 128 * 128; i += 256) {
        int r = i >> 7, c = i & 127;
        Tm[r * 136 + c] = f2bf(tmp[((size_t)b * QQ + r) * DD + c]);
    }
    __builtin_amdgcn_s_wait_asynccnt(0);
#else
    for (int i = tid; i < 128 * 128; i += 256) {
        int r = i >> 7, c = i & 127;
        Aa[r * 136 + c] = act[((size_t)b * CC + c0 + r) * QQ + c];
        Tm[r * 136 + c] = f2bf(tmp[((size_t)b * QQ + r) * DD + c]);
    }
#endif
    __syncthreads();

    const int w = tid >> 5, lane = tid & 31, m0 = w * 16;
    v8f acc[8];
#pragma unroll
    for (int nt = 0; nt < 8; ++nt)
#pragma unroll
        for (int r = 0; r < 8; ++r) acc[nt][r] = 0.f;

    for (int kk = 0; kk < QQ; kk += 32) {              // M=c, K=q, N=d
        v16bf a = ldA(Aa, 136, m0, kk, lane);
#pragma unroll
        for (int nt = 0; nt < 8; ++nt) {
            v16bf bf = ldB_kn(Tm, 136, kk, nt * 16, lane);
            acc[nt] = wmma_bf16(a, bf, acc[nt]);
        }
    }
#pragma unroll
    for (int nt = 0; nt < 8; ++nt) {
        int n = nt * 16 + (lane & 15);
#pragma unroll
        for (int r = 0; r < 8; ++r) {
            int m = m0 + r + ((lane >> 4) << 3);
            q2c_out[((size_t)b * CC + c0 + m) * DD + n] = acc[nt][r];
        }
    }
}

// ---------------------------------------------------------------------------
extern "C" void kernel_launch(void* const* d_in, const int* in_sizes, int n_in,
                              void* d_out, int out_size, void* d_ws, size_t ws_size,
                              hipStream_t stream) {
    const float* xc   = (const float*)d_in[0];            // (B,C,D) f32
    const float* xq   = (const float*)d_in[1];            // (B,Q,D) f32
    const unsigned char* cmask = (const unsigned char*)d_in[2];  // (B,C) bool (1B)
    const unsigned char* qmask = (const unsigned char*)d_in[3];  // (B,Q) bool (1B)
    const float* W0   = (const float*)d_in[4];            // (D,1)
    const float* W1   = (const float*)d_in[5];            // (D,1)
    const float* W2   = (const float*)d_in[6];            // (1,1,D)
    const float* bias = (const float*)d_in[7];            // (1,)

    float* c2q_out = (float*)d_out;                       // (B,C,D)
    float* q2c_out = c2q_out + (size_t)BB * CC * DD;      // (B,C,D)

    // workspace carve: sim f32 (33.5MB) | act bf16 (16.8MB) | colmax | colsum | tmp
    char* wsb = (char*)d_ws;
    float*          sim    = (float*)wsb;
    unsigned short* act    = (unsigned short*)(wsb + (size_t)BB * CC * QQ * 4);
    float*          colmax = (float*)((char*)act + (size_t)BB * CC * QQ * 2);
    float*          colsum = colmax + (size_t)BB * QQ;
    float*          tmp    = colsum + (size_t)BB * QQ;

    const dim3 gridT(CC / 128, BB);
    const size_t sh1 = 2 * 128 * 136 * sizeof(unsigned short) + 256 * sizeof(float) + 256;
    const size_t sh3 = 128 * 132 * sizeof(float) + 2 * 128 * 136 * sizeof(unsigned short);
    const size_t sh4 = 2 * 32 * 136 * sizeof(unsigned short) + 256 * sizeof(float);
    const size_t sh5 = 2 * 128 * 136 * sizeof(unsigned short);

    k_sim<<<gridT, 256, sh1, stream>>>(xc, xq, cmask, qmask, W0, W1, W2, bias, sim);
    k_colstats<<<BB, 1024, 0, stream>>>(sim, colmax, colsum);
    k_rowsoft_c2q<<<gridT, 256, sh3, stream>>>(sim, xq, act, c2q_out);
    k_tmp<<<BB, 256, sh4, stream>>>(sim, xc, colmax, colsum, tmp);
    k_q2c<<<gridT, 256, sh5, stream>>>(act, tmp, q2c_out);
}
